// NaryTreeLSTM_90460601188796
// MI455X (gfx1250) — matrix operand order repaired and checked by reading, not compiled, not run on hardware
//
#include <hip/hip_runtime.h>
#include <hip/hip_bf16.h>

// ---------------------------------------------------------------------------
// Binary TreeLSTM, depth 16, H = D_IN = 256, K = 2 children.
// Heap indexing: children of global node g are 2g+1, 2g+2.
// Per level (bottom-up): 5 fused GEMMs per node batch via bf16 WMMA.
// Block = 512 threads (16 waves) handles 32 nodes (two 16-row M tiles);
// wave w owns output columns [16w, 16w+16) and both M tiles, so every
// weight (B) fragment feeds two WMMAs -> halves L2 weight traffic.
// ---------------------------------------------------------------------------

typedef __attribute__((ext_vector_type(16))) __bf16 v16bf;
typedef __attribute__((ext_vector_type(8)))  float  v8f;
typedef __attribute__((ext_vector_type(4)))  int    v4i;

union FragB16 { v4i i[2]; v16bf v; };

static __device__ __forceinline__ unsigned short f32_to_bf16_rne(float f) {
    unsigned int u = __float_as_uint(f);
    u += 0x7FFFu + ((u >> 16) & 1u);   // round-to-nearest-even
    return (unsigned short)(u >> 16);
}

static __device__ __forceinline__ float sigmoidf_(float x) {
    return 1.0f / (1.0f + __expf(-x));
}

// Load a 16-element bf16 fragment as two 16-byte chunks; chunk starts are
// 16 elements apart for both A and B fragments (`step` is ignored).
static __device__ __forceinline__ v16bf load2(const unsigned short* p, int step) {
    FragB16 f;
    (void)step;
    f.i[0] = *(const v4i*)(p);
    f.i[1] = *(const v4i*)(p + 16);
    return f.v;
}

static __device__ __forceinline__ v8f wmma_bf16(v16bf a, v16bf b, v8f c) {
    // (neg_a, A, neg_b, B, c_mod, C, reuse_a, reuse_b)
    return __builtin_amdgcn_wmma_f32_16x16x32_bf16(
        false, a, false, b, (short)0, c, false, false);
}

// ---------------------------------------------------------------------------
// fp32 -> bf16 weight conversion (keeps [out][in] layout; B-fragment friendly)
// ---------------------------------------------------------------------------
__global__ void convert_f32_to_bf16(const float* __restrict__ src,
                                    unsigned short* __restrict__ dst, int n) {
    int i = blockIdx.x * blockDim.x + threadIdx.x;
    if (i < n) dst[i] = f32_to_bf16_rne(src[i]);
}

// ---------------------------------------------------------------------------
// One tree level.
// ---------------------------------------------------------------------------
__global__ __launch_bounds__(512)
void tree_lstm_level(const float* __restrict__ emb,
                     const unsigned short* __restrict__ Wb,   // [4][256][256] (i,o,u,f)
                     const unsigned short* __restrict__ Ub,   // [4][2][256][256]
                     const float* __restrict__ b_i,
                     const float* __restrict__ b_o,
                     const float* __restrict__ b_u,
                     const float* __restrict__ b_f,
                     unsigned short* __restrict__ h_bf,       // [65536][256] bf16
                     float* __restrict__ c_buf,               // [65536][256] f32
                     int node_base, int n_nodes, int is_leaf,
                     float* __restrict__ out_root)            // null unless root level
{
    constexpr int HD = 256;
    __shared__ __align__(16) unsigned short Xs [32 * HD];   // 16 KB
    __shared__ __align__(16) unsigned short H0s[32 * HD];   // 16 KB
    __shared__ __align__(16) unsigned short H1s[32 * HD];   // 16 KB

    const int tid  = threadIdx.x;
    const int lane = tid & 31;
    const int wave = tid >> 5;
    const int row0 = blockIdx.x * 32;

    // ---- stage X (fp32->bf16) and child hidden states (bf16) into LDS ----
    for (int idx = tid; idx < 32 * HD; idx += 512) {
        const int m   = idx >> 8;
        const int col = idx & 255;
        const int r   = row0 + m;
        float xv = 0.0f;
        unsigned short h0 = 0, h1 = 0;
        if (r < n_nodes) {
            const int node = node_base + r;
            xv = emb[node * HD + col];
            if (!is_leaf) {
                h0 = h_bf[(2 * node + 1) * HD + col];
                h1 = h_bf[(2 * node + 2) * HD + col];
            }
        }
        Xs[idx]  = f32_to_bf16_rne(xv);
        H0s[idx] = h0;
        H1s[idx] = h1;
    }
    __syncthreads();

    // ---- per-wave WMMA fragment addressing ----
    // A fragment (16x32, bf16): lane -> row M = lane&15;
    //   elements 0..7  = K + (lane>>4)*8 + {0..7}
    //   elements 8..15 = K + 16 + (lane>>4)*8 + {0..7}
    const int aRow = lane & 15;
    const int aOff = (lane >> 4) * 8;
    // B fragment (32x16, bf16): lane -> column N = lane&15 (= weight row d);
    //   elements 0..7 = K + (lane>>4)*8 + {0..7}, 8..15 = K + 16 + (lane>>4)*8 + {0..7}
    const int colbase = wave * 16;
    const int bRow    = colbase + (lane & 15);
    const int bOff    = (lane >> 4) * 8;

    // ---- accumulators: [M tile][gate i,o,u,f0,f1], seeded with biases ----
    v8f acc[2][5];
    {
        const int col = colbase + (lane & 15);
        const float vi = b_i[col], vo = b_o[col], vu = b_u[col], vf = b_f[col];
        #pragma unroll
        for (int t = 0; t < 2; ++t)
            #pragma unroll
            for (int e = 0; e < 8; ++e) {
                acc[t][0][e] = vi; acc[t][1][e] = vo; acc[t][2][e] = vu;
                acc[t][3][e] = vf; acc[t][4][e] = vf;
            }
    }

    const unsigned short* Wi  = Wb + 0 * 65536;
    const unsigned short* Wo  = Wb + 1 * 65536;
    const unsigned short* Wu  = Wb + 2 * 65536;
    const unsigned short* Wf  = Wb + 3 * 65536;
    const unsigned short* Ui0 = Ub + 0 * 65536;
    const unsigned short* Ui1 = Ub + 1 * 65536;
    const unsigned short* Uo0 = Ub + 2 * 65536;
    const unsigned short* Uo1 = Ub + 3 * 65536;
    const unsigned short* Uu0 = Ub + 4 * 65536;
    const unsigned short* Uu1 = Ub + 5 * 65536;
    const unsigned short* Uf0 = Ub + 6 * 65536;
    const unsigned short* Uf1 = Ub + 7 * 65536;

    const int aBase0 = aRow * HD + aOff;          // M tile 0
    const int aBase1 = (16 + aRow) * HD + aOff;   // M tile 1
    const int wBase  = bRow * HD + bOff;

    // ---- main GEMM loop: K = 256 in steps of 32; 26 WMMAs / 12 B-loads ----
    for (int kb = 0; kb < HD; kb += 32) {
        const int wOffs = wBase + kb;
        const v16bf ax0 = load2(&Xs[aBase0 + kb], 16);
        const v16bf ax1 = load2(&Xs[aBase1 + kb], 16);
        {   // input projections: gates i, o, u
            const v16bf b0 = load2(Wi + wOffs, 8);
            acc[0][0] = wmma_bf16(ax0, b0, acc[0][0]);
            acc[1][0] = wmma_bf16(ax1, b0, acc[1][0]);
            const v16bf b1 = load2(Wo + wOffs, 8);
            acc[0][1] = wmma_bf16(ax0, b1, acc[0][1]);
            acc[1][1] = wmma_bf16(ax1, b1, acc[1][1]);
            const v16bf b2 = load2(Wu + wOffs, 8);
            acc[0][2] = wmma_bf16(ax0, b2, acc[0][2]);
            acc[1][2] = wmma_bf16(ax1, b2, acc[1][2]);
            // forget gate input projection feeds both f0 and f1
            const v16bf b3 = load2(Wf + wOffs, 8);
            acc[0][3] = wmma_bf16(ax0, b3, acc[0][3]);
            acc[0][4] = wmma_bf16(ax0, b3, acc[0][4]);
            acc[1][3] = wmma_bf16(ax1, b3, acc[1][3]);
            acc[1][4] = wmma_bf16(ax1, b3, acc[1][4]);
        }
        if (!is_leaf) {
            const v16bf ah0_0 = load2(&H0s[aBase0 + kb], 16);
            const v16bf ah0_1 = load2(&H0s[aBase1 + kb], 16);
            const v16bf ah1_0 = load2(&H1s[aBase0 + kb], 16);
            const v16bf ah1_1 = load2(&H1s[aBase1 + kb], 16);
            {   // child-0 recurrent projections
                const v16bf b0 = load2(Ui0 + wOffs, 8);
                acc[0][0] = wmma_bf16(ah0_0, b0, acc[0][0]);
                acc[1][0] = wmma_bf16(ah0_1, b0, acc[1][0]);
                const v16bf b1 = load2(Uo0 + wOffs, 8);
                acc[0][1] = wmma_bf16(ah0_0, b1, acc[0][1]);
                acc[1][1] = wmma_bf16(ah0_1, b1, acc[1][1]);
                const v16bf b2 = load2(Uu0 + wOffs, 8);
                acc[0][2] = wmma_bf16(ah0_0, b2, acc[0][2]);
                acc[1][2] = wmma_bf16(ah0_1, b2, acc[1][2]);
                const v16bf b3 = load2(Uf0 + wOffs, 8);
                acc[0][3] = wmma_bf16(ah0_0, b3, acc[0][3]);
                acc[1][3] = wmma_bf16(ah0_1, b3, acc[1][3]);
            }
            {   // child-1 recurrent projections
                const v16bf b0 = load2(Ui1 + wOffs, 8);
                acc[0][0] = wmma_bf16(ah1_0, b0, acc[0][0]);
                acc[1][0] = wmma_bf16(ah1_1, b0, acc[1][0]);
                const v16bf b1 = load2(Uo1 + wOffs, 8);
                acc[0][1] = wmma_bf16(ah1_0, b1, acc[0][1]);
                acc[1][1] = wmma_bf16(ah1_1, b1, acc[1][1]);
                const v16bf b2 = load2(Uu1 + wOffs, 8);
                acc[0][2] = wmma_bf16(ah1_0, b2, acc[0][2]);
                acc[1][2] = wmma_bf16(ah1_1, b2, acc[1][2]);
                const v16bf b3 = load2(Uf1 + wOffs, 8);
                acc[0][4] = wmma_bf16(ah1_0, b3, acc[0][4]);
                acc[1][4] = wmma_bf16(ah1_1, b3, acc[1][4]);
            }
        }
    }

    // ---- epilogue: activations, cell update, stores ----
    // Accumulator layout: lane l, vreg v -> M = (l>>4)*8 + v (within tile),
    // N = l&15.
    const int col = colbase + (lane & 15);
    #pragma unroll
    for (int t = 0; t < 2; ++t) {
        #pragma unroll
        for (int v = 0; v < 8; ++v) {
            const int m = t * 16 + ((lane >> 4) * 8) + v;
            if (row0 + m >= n_nodes) continue;
            const int node = node_base + row0 + m;
            const float iv = sigmoidf_(acc[t][0][v]);
            const float ov = sigmoidf_(acc[t][1][v]);
            const float uv = tanhf(acc[t][2][v]);
            float cc = iv * uv;
            if (!is_leaf) {
                const float f0 = sigmoidf_(acc[t][3][v]);
                const float f1 = sigmoidf_(acc[t][4][v]);
                cc += f0 * c_buf[(2 * node + 1) * HD + col]
                    + f1 * c_buf[(2 * node + 2) * HD + col];
            }
            const float hh = ov * tanhf(cc);
            h_bf[node * HD + col] = f32_to_bf16_rne(hh);
            c_buf[node * HD + col] = cc;
            if (out_root && node == 0) {
                out_root[col]      = hh;   // h[0]
                out_root[HD + col] = cc;   // c[0]
            }
        }
    }
}

// ---------------------------------------------------------------------------
// Host-side launch: weight conversion once per call, then 16 level launches
// (stream ordering provides the level-to-level dependency).
// ---------------------------------------------------------------------------
extern "C" void kernel_launch(void* const* d_in, const int* in_sizes, int n_in,
                              void* d_out, int out_size, void* d_ws, size_t ws_size,
                              hipStream_t stream) {
    (void)in_sizes; (void)n_in; (void)out_size; (void)ws_size;

    const float* emb = (const float*)d_in[0];
    const float* W_i = (const float*)d_in[1];
    const float* b_i = (const float*)d_in[2];
    const float* U_i = (const float*)d_in[3];
    const float* W_o = (const float*)d_in[4];
    const float* b_o = (const float*)d_in[5];
    const float* U_o = (const float*)d_in[6];
    const float* W_u = (const float*)d_in[7];
    const float* b_u = (const float*)d_in[8];
    const float* U_u = (const float*)d_in[9];
    const float* W_f = (const float*)d_in[10];
    const float* b_f = (const float*)d_in[11];
    const float* U_f = (const float*)d_in[12];

    // workspace layout
    char* ws = (char*)d_ws;
    unsigned short* Wb   = (unsigned short*)(ws);                       // 4*65536 bf16   (512 KB)
    unsigned short* Ub   = (unsigned short*)(ws + 524288);              // 8*65536 bf16   (1 MB)
    unsigned short* h_bf = (unsigned short*)(ws + 524288 + 1048576);    // 65536*256 bf16 (32 MB)
    float*          c_bf = (float*)(ws + 524288 + 1048576 + 33554432);  // 65536*256 f32  (64 MB)

    const int NW = 256 * 256;        // 65536
    const int NU = 2 * 256 * 256;    // 131072

    convert_f32_to_bf16<<<NW / 256, 256, 0, stream>>>(W_i, Wb + 0 * NW, NW);
    convert_f32_to_bf16<<<NW / 256, 256, 0, stream>>>(W_o, Wb + 1 * NW, NW);
    convert_f32_to_bf16<<<NW / 256, 256, 0, stream>>>(W_u, Wb + 2 * NW, NW);
    convert_f32_to_bf16<<<NW / 256, 256, 0, stream>>>(W_f, Wb + 3 * NW, NW);
    convert_f32_to_bf16<<<NU / 256, 256, 0, stream>>>(U_i, Ub + 0 * NU, NU);
    convert_f32_to_bf16<<<NU / 256, 256, 0, stream>>>(U_o, Ub + 1 * NU, NU);
    convert_f32_to_bf16<<<NU / 256, 256, 0, stream>>>(U_u, Ub + 2 * NU, NU);
    convert_f32_to_bf16<<<NU / 256, 256, 0, stream>>>(U_f, Ub + 3 * NU, NU);

    // bottom-up over levels 15 .. 0
    for (int l = 15; l >= 0; --l) {
        const int n    = 1 << l;
        const int base = (1 << l) - 1;
        const int blocks = (n + 31) / 32;
        tree_lstm_level<<<blocks, 512, 0, stream>>>(
            emb, Wb, Ub, b_i, b_o, b_u, b_f, h_bf, c_bf,
            base, n, (l == 15) ? 1 : 0,
            (l == 0) ? (float*)d_out : nullptr);
    }
}
